// TrajectoryPredictor_1992864825878
// MI455X (gfx1250) — matrix operand order repaired and checked by reading, not compiled, not run on hardware
//
#include <hip/hip_runtime.h>
#include <math.h>

// ---------------- types ----------------
typedef __attribute__((ext_vector_type(16))) __bf16    v16bf;
typedef __attribute__((ext_vector_type(8)))  float     v8f;
typedef __attribute__((ext_vector_type(4)))  unsigned  v4u;

#define BB   4
#define NN   2048
#define TT   20
#define NNODE (BB*NN)          // 8192
#define LH   128
#define G4   512               // 4*LH
#define KDIM 168               // padded stride: 128 h + 18 x + pad (bank-conflict pad)
#define ROWS 32                // sequences per block (2 M-tiles of 16)

union Frag { unsigned u[8]; v4u q[2]; v16bf v; };

__device__ __forceinline__ unsigned short f2bf(float f) {
    unsigned u = __float_as_uint(f);
    unsigned r = u + 0x7FFFu + ((u >> 16) & 1u);   // round-to-nearest-even
    return (unsigned short)(r >> 16);
}
__device__ __forceinline__ float bf2f(unsigned short h) {
    return __uint_as_float(((unsigned)h) << 16);
}
__device__ __forceinline__ float sigmoidf(float x) {
    return 1.0f / (1.0f + __expf(-x));
}

// ---------------------------------------------------------------------------
// Kernel 1: per-node GAT features (Wh0, Wh1, s, t), beam MLP, conf, corr_needed
// ---------------------------------------------------------------------------
__global__ __launch_bounds__(256)
void k_node(const float* __restrict__ traj,   // (B,N,T,2)
            const float* __restrict__ env,    // (B,N,16)
            const float* __restrict__ gW,     // (2,64)
            const float* __restrict__ gA,     // (128,)
            const float* __restrict__ bW1,    // (16,64)
            const float* __restrict__ bB1,    // (64,)
            const float* __restrict__ bW2,    // (64,2)
            const float* __restrict__ bB2,    // (2,)
            float* __restrict__ wh0, float* __restrict__ wh1,
            float* __restrict__ ss,  float* __restrict__ tt,
            float* __restrict__ beam, float* __restrict__ corr,
            float* __restrict__ out)          // conf -> out[16384 + node]
{
    int node = blockIdx.x * 256 + threadIdx.x;
    if (node >= NNODE) return;

    float x0 = traj[(node * TT + (TT - 1)) * 2 + 0];
    float x1 = traj[(node * TT + (TT - 1)) * 2 + 1];

    float dI0 = 0.f, dI1 = 0.f, dJ0 = 0.f, dJ1 = 0.f;
    #pragma unroll
    for (int f = 0; f < 64; ++f) {
        float w0 = gW[f], w1 = gW[64 + f];
        float ai = gA[f], aj = gA[64 + f];
        dI0 += w0 * ai; dI1 += w1 * ai;
        dJ0 += w0 * aj; dJ1 += w1 * aj;
    }
    ss[node] = x0 * dI0 + x1 * dI1;
    tt[node] = x0 * dJ0 + x1 * dJ1;
    wh0[node] = x0 * gW[0] + x1 * gW[64];
    wh1[node] = x0 * gW[1] + x1 * gW[65];

    float ef[16];
    #pragma unroll
    for (int i = 0; i < 16; ++i) ef[i] = env[node * 16 + i];
    float bo0 = bB2[0], bo1 = bB2[1];
    #pragma unroll 4
    for (int j = 0; j < 64; ++j) {
        float h = bB1[j];
        #pragma unroll
        for (int i = 0; i < 16; ++i) h += ef[i] * bW1[i * 64 + j];
        h = fmaxf(h, 0.f);
        bo0 += h * bW2[j * 2 + 0];
        bo1 += h * bW2[j * 2 + 1];
    }
    beam[node * 2 + 0] = bo0;
    beam[node * 2 + 1] = bo1;

    float mag  = sqrtf(bo0 * bo0 + bo1 * bo1);
    float conf = 1.0f / (1.0f + __expf(mag));       // sigmoid(-mag)
    corr[node] = (conf < 0.5f) ? 1.0f : 0.0f;
    out[NNODE * 2 + node] = conf;
}

// ---------------------------------------------------------------------------
// Kernel 2: GAT row softmax (only features 0,1 of gat_out are consumed)
// ---------------------------------------------------------------------------
__global__ __launch_bounds__(256)
void k_gat_row(const float* __restrict__ wh0, const float* __restrict__ wh1,
               const float* __restrict__ ss,  const float* __restrict__ tt,
               const float* __restrict__ beam,
               float* __restrict__ first_pred)
{
    __shared__ float sm[256], sZ[256], s0[256], s1[256];
    int row = blockIdx.x;                 // 0..8191
    int jb  = (row >> 11) << 11;          // batch base
    int tid = threadIdx.x;
    float si = ss[row];

    float m = -INFINITY, Z = 0.f, n0 = 0.f, n1 = 0.f;
    for (int j = tid; j < NN; j += 256) {
        float e = si + tt[jb + j];
        e = (e > 0.f) ? e : 0.01f * e;            // leaky_relu
        if (e > m) {
            float sc = __expf(m - e);
            Z *= sc; n0 *= sc; n1 *= sc; m = e;
        }
        float w = __expf(e - m);
        Z  += w;
        n0 += w * wh0[jb + j];
        n1 += w * wh1[jb + j];
    }
    sm[tid] = m; sZ[tid] = Z; s0[tid] = n0; s1[tid] = n1;
    for (int off = 128; off > 0; off >>= 1) {
        __syncthreads();
        if (tid < off) {
            float ma = sm[tid], mb = sm[tid + off];
            float M  = fmaxf(ma, mb);
            float sa = __expf(ma - M), sb = __expf(mb - M);
            sm[tid] = M;
            sZ[tid] = sZ[tid] * sa + sZ[tid + off] * sb;
            s0[tid] = s0[tid] * sa + s0[tid + off] * sb;
            s1[tid] = s1[tid] * sa + s1[tid + off] * sb;
        }
    }
    __syncthreads();
    if (tid == 0) {
        float inv = 1.0f / sZ[0];
        first_pred[row * 2 + 0] = s0[0] * inv + beam[row * 2 + 0];
        first_pred[row * 2 + 1] = s1[0] * inv + beam[row * 2 + 1];
    }
}

// ---------------------------------------------------------------------------
// Kernel 3: fused LSTM via bf16 WMMA.
//   K = [h(128) | x(18) | pad] ; W_cat = [Whh | Wih | 0] (512 x KDIM) bf16 in LDS.
//   Block = 32 sequences (2 M-tiles); 8 waves; wave w owns hidden slice
//   [16w, 16w+16) and computes all 4 gates for it (B fragments reused across
//   both M-tiles). Cell state in VGPRs; h ping-pongs through shared LDS with
//   two barriers per step. KDIM=168 pads LDS rows to kill 4-way bank conflicts.
// ---------------------------------------------------------------------------
__global__ __launch_bounds__(256)
void k_lstm(const float* __restrict__ traj,   // (B,N,T,2)
            const float* __restrict__ env,    // (B,N,16)
            const float* __restrict__ Wih,    // (512,18)
            const float* __restrict__ Whh,    // (512,128)
            const float* __restrict__ bih,    // (512,)
            const float* __restrict__ bhh,    // (512,)
            const float* __restrict__ fcW,    // (128,2)
            const float* __restrict__ fcb,    // (2,)
            const float* __restrict__ first_pred,
            const float* __restrict__ corr_needed,
            float* __restrict__ out)
{
    extern __shared__ char smem[];
    unsigned short* Wc = (unsigned short*)smem;                       // 512*KDIM bf16
    unsigned short* HX = (unsigned short*)(smem + G4 * KDIM * 2);     // ROWS*KDIM bf16
    float*          BS = (float*)(smem + (G4 + ROWS) * KDIM * 2);     // 512 f32 bias

    const int tid = threadIdx.x;
    const int rowbase = blockIdx.x * ROWS;

    // ---- stage W_cat = [Whh | Wih | 0] into LDS (bf16) ----
    for (int idx = tid; idx < G4 * KDIM; idx += 256) {
        int n = idx / KDIM, k = idx - n * KDIM;
        float v = 0.f;
        if (k < 128)      v = Whh[n * 128 + k];
        else if (k < 146) v = Wih[n * 18 + (k - 128)];
        Wc[idx] = f2bf(v);
    }
    // ---- init [h|x] rows: h=0, x[2..17]=env (constant over t) ----
    for (int idx = tid; idx < ROWS * KDIM; idx += 256) {
        int r = idx / KDIM, k = idx - r * KDIM;
        float v = 0.f;
        if (k >= 130 && k < 146) v = env[(rowbase + r) * 16 + (k - 130)];
        HX[idx] = f2bf(v);
    }
    // ---- bias sums ----
    for (int idx = tid; idx < G4; idx += 256) BS[idx] = bih[idx] + bhh[idx];
    __syncthreads();

    const int wave = tid >> 5, lane = tid & 31;   // wave == hidden tile ht
    const int mrow   = lane & 15;                 // A row / B col / C col
    const int hiHalf = (lane >> 4);               // 0 or 1
    const int khalfA = hiHalf * 8;
    const int khalfB = hiHalf * 16;

    float cst[2][8];                              // cell state (2 M-tiles, C layout)
    #pragma unroll
    for (int mt = 0; mt < 2; ++mt)
        #pragma unroll
        for (int r = 0; r < 8; ++r) cst[mt][r] = 0.f;

    for (int t = 0; t < TT; ++t) {
        // wave 0 writes this step's traj inputs into x-slot (K=128,129), 1 row/lane
        if (wave == 0) {
            int node = rowbase + lane;
            float a = traj[(node * TT + t) * 2 + 0];
            float b = traj[(node * TT + t) * 2 + 1];
            unsigned pack = (unsigned)f2bf(a) | ((unsigned)f2bf(b) << 16);
            *(unsigned*)&HX[lane * KDIM + 128] = pack;
        }
        __syncthreads();   // h(t-1) + x(t) visible to all waves

        // A fragments: 2 M-tiles x 5 K-chunks, each as 2x ds_load_b128
        Frag afr[2][5];
        #pragma unroll
        for (int mt = 0; mt < 2; ++mt) {
            const unsigned short* arow = HX + (mt * 16 + mrow) * KDIM;
            #pragma unroll
            for (int kc = 0; kc < 5; ++kc) {
                int k0 = kc * 32 + khalfA;
                afr[mt][kc].q[0] = *(const v4u*)(arow + k0);
                afr[mt][kc].q[1] = *(const v4u*)(arow + k0 + 16);
            }
        }
        __syncthreads();   // all A reads done before h(t) overwrites HX

        v8f acc[2][4];
        #pragma unroll
        for (int g = 0; g < 4; ++g) {
            int nb  = g * 128 + wave * 16;        // gate row block in W_cat
            float bsum = BS[nb + mrow];
            acc[0][g] = (v8f){bsum, bsum, bsum, bsum, bsum, bsum, bsum, bsum};
            acc[1][g] = acc[0][g];
            const unsigned short* brow = Wc + (nb + mrow) * KDIM;
            #pragma unroll
            for (int kc = 0; kc < 5; ++kc) {
                Frag bfr;                          // one B fragment, two WMMAs
                int k0 = kc * 32 + khalfA;
                bfr.q[0] = *(const v4u*)(brow + k0);
                bfr.q[1] = *(const v4u*)(brow + k0 + 16);
                acc[0][g] = __builtin_amdgcn_wmma_f32_16x16x32_bf16(
                    false, afr[0][kc].v, false, bfr.v, (short)0, acc[0][g],
                    false, false);
                acc[1][g] = __builtin_amdgcn_wmma_f32_16x16x32_bf16(
                    false, afr[1][kc].v, false, bfr.v, (short)0, acc[1][g],
                    false, false);
            }
        }
        // gate nonlinearities (C layout), write h slice [16*wave, 16*wave+16)
        #pragma unroll
        for (int mt = 0; mt < 2; ++mt) {
            #pragma unroll
            for (int r = 0; r < 8; ++r) {
                float iv = acc[mt][0][r], fv = acc[mt][1][r];
                float gv = acc[mt][2][r], ov = acc[mt][3][r];
                float c  = sigmoidf(fv) * cst[mt][r] + sigmoidf(iv) * tanhf(gv);
                cst[mt][r] = c;
                float h  = sigmoidf(ov) * tanhf(c);
                int m = mt * 16 + r + 8 * hiHalf;
                HX[m * KDIM + wave * 16 + mrow] = f2bf(h);
            }
        }
    }
    __syncthreads();       // final h complete

    // ---- epilogue: correction = hT @ fcW + fcb ; fused final output ----
    if (tid < ROWS) {
        int node = rowbase + tid;
        float c0 = fcb[0], c1 = fcb[1];
        for (int k = 0; k < LH; ++k) {
            float h = bf2f(HX[tid * KDIM + k]);
            c0 += h * fcW[k * 2 + 0];
            c1 += h * fcW[k * 2 + 1];
        }
        float cn = corr_needed[node];
        out[node * 2 + 0] = first_pred[node * 2 + 0] + c0 * cn;
        out[node * 2 + 1] = first_pred[node * 2 + 1] + c1 * cn;
    }
}

// ---------------------------------------------------------------------------
extern "C" void kernel_launch(void* const* d_in, const int* in_sizes, int n_in,
                              void* d_out, int out_size, void* d_ws, size_t ws_size,
                              hipStream_t stream) {
    const float* traj = (const float*)d_in[0];
    const float* env  = (const float*)d_in[1];
    const float* gW   = (const float*)d_in[2];
    const float* gA   = (const float*)d_in[3];
    const float* bW1  = (const float*)d_in[4];
    const float* bB1  = (const float*)d_in[5];
    const float* bW2  = (const float*)d_in[6];
    const float* bB2  = (const float*)d_in[7];
    const float* Wih  = (const float*)d_in[8];
    const float* Whh  = (const float*)d_in[9];
    const float* bih  = (const float*)d_in[10];
    const float* bhh  = (const float*)d_in[11];
    const float* fcW  = (const float*)d_in[12];
    const float* fcb  = (const float*)d_in[13];
    float* out = (float*)d_out;

    // workspace layout (floats)
    float* ws   = (float*)d_ws;
    float* wh0  = ws;                  // 8192
    float* wh1  = wh0 + NNODE;         // 8192
    float* ssb  = wh1 + NNODE;         // 8192
    float* ttb  = ssb + NNODE;         // 8192
    float* beam = ttb + NNODE;         // 16384
    float* corr = beam + NNODE * 2;    // 8192
    float* fp   = corr + NNODE;        // 16384

    k_node<<<NNODE / 256, 256, 0, stream>>>(traj, env, gW, gA, bW1, bB1, bW2,
                                            bB2, wh0, wh1, ssb, ttb, beam,
                                            corr, out);
    k_gat_row<<<NNODE, 256, 0, stream>>>(wh0, wh1, ssb, ttb, beam, fp);

    size_t lds = (size_t)(G4 + ROWS) * KDIM * 2 + (size_t)G4 * 4;  // 184,832 B
    k_lstm<<<NNODE / ROWS, 256, lds, stream>>>(traj, env, Wih, Whh, bih, bhh,
                                               fcW, fcb, fp, corr, out);
}
